// scGNN_9225589751938
// MI455X (gfx1250) — matrix operand order repaired and checked by reading, not compiled, not run on hardware
//
#include <hip/hip_runtime.h>

typedef __attribute__((ext_vector_type(16))) __bf16 v16bf;
typedef __attribute__((ext_vector_type(8)))  float  v8f;

namespace {
constexpr int NN   = 20000;   // nodes
constexpr int EE   = 320000;  // edges
constexpr int DIN  = 2000;
constexpr int DMID = 512;
constexpr int KP1  = 2016;    // DIN padded to multiple of 32 (K of GEMM1)
constexpr int NP2  = 2048;    // DIN padded to multiple of 64 (N of GEMM2)

constexpr size_t algn(size_t x) { return (x + 255) & ~size_t(255); }
constexpr size_t OFF_DEG  = 0;
constexpr size_t OFF_DINV = algn(OFF_DEG  + (size_t)NN * 4);
constexpr size_t OFF_HBF  = algn(OFF_DINV + (size_t)NN * 4);
constexpr size_t OFF_WT2  = algn(OFF_HBF  + (size_t)NN * DMID * 2);
constexpr size_t OFF_REG  = algn(OFF_WT2  + (size_t)NP2 * DMID * 2);
// phase-1 region (x_bf16 | Wt1 | h0 | agg1) is dead once h_bf16 exists,
// so h2 (20000x2048 f32) aliases it.
constexpr size_t OFF_XBF  = OFF_REG;
constexpr size_t OFF_WT1  = algn(OFF_XBF + (size_t)NN * KP1 * 2);
constexpr size_t OFF_H0   = algn(OFF_WT1 + (size_t)DMID * KP1 * 2);
constexpr size_t OFF_AGG1 = algn(OFF_H0  + (size_t)NN * DMID * 4);
constexpr size_t OFF_H2   = OFF_REG;
}

__device__ __forceinline__ unsigned short f2bf(float f) {
  unsigned int u = __float_as_uint(f);
  u += 0x7fffu + ((u >> 16) & 1u);     // round-to-nearest-even
  return (unsigned short)(u >> 16);
}

// ---------------- degree / norm ----------------
__global__ void k_init_deg(float* deg, int n) {
  int i = blockIdx.x * blockDim.x + threadIdx.x;
  if (i < n) deg[i] = 1.0f;            // self-loop contribution
}
__global__ void k_deg(const long long* __restrict__ dst, float* deg, int e) {
  int i = blockIdx.x * blockDim.x + threadIdx.x;
  if (i < e) atomicAdd(&deg[(int)dst[i]], 1.0f);
}
__global__ void k_dinv(const float* __restrict__ deg, float* dinv, int n) {
  int i = blockIdx.x * blockDim.x + threadIdx.x;
  if (i < n) dinv[i] = rsqrtf(deg[i]); // deg >= 1 always
}

// ---------------- casts / transposes ----------------
__global__ void k_cvt_pad(const float* __restrict__ src, unsigned short* __restrict__ dst,
                          int cols, int colsPad, int total) {
  int idx = blockIdx.x * blockDim.x + threadIdx.x;
  if (idx >= total) return;
  int r = idx / colsPad, c = idx - r * colsPad;
  dst[idx] = (c < cols) ? f2bf(src[(size_t)r * cols + c]) : (unsigned short)0;
}
// Wt[n*KPad + k] = bf16(W[k*Ncols + n]), zero-padded in both dims
__global__ void k_cvt_transpose(const float* __restrict__ W, unsigned short* __restrict__ Wt,
                                int K, int Ncols, int KPad, int total) {
  int idx = blockIdx.x * blockDim.x + threadIdx.x;
  if (idx >= total) return;
  int n = idx / KPad, k = idx - n * KPad;
  Wt[idx] = (k < K && n < Ncols) ? f2bf(W[(size_t)k * Ncols + n]) : (unsigned short)0;
}

// ---------------- WMMA bf16 GEMM: C[M,N] = A[M,Kp] * Bt[N,Kp]^T ----------------
// One wave computes a 32x64 strip: 2 M-tiles x 4 N-tiles = 8 accumulators.
// Software-pipelined: next K-step fragments are loaded before current WMMAs,
// so s_wait_loadcnt tolerates in-flight prefetches and L2 latency overlaps math.
// Fragment packing per CDNA5 ISA 7.12.2 (wave32):
//   A 16x32 bf16 : lanes 0-15 = rows, elems 0..7 -> K k0..k0+7, 8..15 -> K k0+16..k0+23
//                  lanes 16-31: elems 0..7 -> K k0+8..k0+15, 8..15 -> K k0+24..k0+31
//   B 32x16 bf16 : same K packing as A (lanes 0-15: k0..k0+7, k0+16..k0+23; lanes 16-31: k0+8..k0+15, k0+24..k0+31)
//   D 16x16 f32  : VGPR r -> row r (lanes 0-15) / row r+8 (lanes 16-31), N = lane%16
typedef union { v16bf bf; uint4 u[2]; } Frag;

#define WMMA_BF16(accv, af, bfg)                                              \
  accv = __builtin_amdgcn_wmma_f32_16x16x32_bf16(false, (af).bf, false,        \
                                                 (bfg).bf, (short)0, accv,     \
                                                 false, false)

__global__ __launch_bounds__(256) void k_gemm_bf16(
    const unsigned short* __restrict__ A, const unsigned short* __restrict__ Bt,
    float* __restrict__ C, int N, int Kp, int nTileN4) {
  const int lane = threadIdx.x & 31;
  const int wave = threadIdx.x >> 5;
  const int gw   = blockIdx.x * 8 + wave;
  const int tm   = gw / nTileN4;                 // 32-row strip
  const int tn0  = (gw - tm * nTileN4) * 4;      // 4 x 16-col tiles

  const int half = (lane >> 4) & 1;
  const int l15  = lane & 15;

  const unsigned short* aR0 = A + (size_t)(tm * 32 + l15) * Kp;
  const unsigned short* aR1 = aR0 + (size_t)16 * Kp;
  const unsigned short* bR[4];
#pragma unroll
  for (int t = 0; t < 4; ++t)
    bR[t] = Bt + (size_t)((tn0 + t) * 16 + l15) * Kp;

  const int kaOff = half ? 8 : 0;
  const int kbOff = half ? 8 : 0;

  auto loadA = [&](Frag& f, const unsigned short* base, int k0) {
    f.u[0] = *(const uint4*)(base + k0 + kaOff);
    f.u[1] = *(const uint4*)(base + k0 + 16 + kaOff);
  };
  auto loadB = [&](Frag& f, const unsigned short* base, int k0) {
    f.u[0] = *(const uint4*)(base + k0 + kbOff);
    f.u[1] = *(const uint4*)(base + k0 + 16 + kbOff);
  };

  v8f acc[2][4] = {};
  Frag a0, a1, b0, b1, b2, b3;
  loadA(a0, aR0, 0); loadA(a1, aR1, 0);
  loadB(b0, bR[0], 0); loadB(b1, bR[1], 0);
  loadB(b2, bR[2], 0); loadB(b3, bR[3], 0);

  for (int k0 = 32; k0 < Kp; k0 += 32) {
    Frag na0, na1, nb0, nb1, nb2, nb3;
    // issue next K-step loads first (prefetch into regs)
    loadA(na0, aR0, k0); loadA(na1, aR1, k0);
    loadB(nb0, bR[0], k0); loadB(nb1, bR[1], k0);
    loadB(nb2, bR[2], k0); loadB(nb3, bR[3], k0);
    // compute current K-step (waits only on last iteration's loads)
    WMMA_BF16(acc[0][0], a0, b0); WMMA_BF16(acc[1][0], a1, b0);
    WMMA_BF16(acc[0][1], a0, b1); WMMA_BF16(acc[1][1], a1, b1);
    WMMA_BF16(acc[0][2], a0, b2); WMMA_BF16(acc[1][2], a1, b2);
    WMMA_BF16(acc[0][3], a0, b3); WMMA_BF16(acc[1][3], a1, b3);
    a0 = na0; a1 = na1; b0 = nb0; b1 = nb1; b2 = nb2; b3 = nb3;
  }
  WMMA_BF16(acc[0][0], a0, b0); WMMA_BF16(acc[1][0], a1, b0);
  WMMA_BF16(acc[0][1], a0, b1); WMMA_BF16(acc[1][1], a1, b1);
  WMMA_BF16(acc[0][2], a0, b2); WMMA_BF16(acc[1][2], a1, b2);
  WMMA_BF16(acc[0][3], a0, b3); WMMA_BF16(acc[1][3], a1, b3);

#pragma unroll
  for (int mt = 0; mt < 2; ++mt) {
    const int rowC = tm * 32 + mt * 16 + (half ? 8 : 0);
#pragma unroll
    for (int t = 0; t < 4; ++t) {
      float* cptr = C + (size_t)rowC * N + (tn0 + t) * 16 + l15;
#pragma unroll
      for (int r = 0; r < 8; ++r)
        cptr[(size_t)r * N] = acc[mt][t][r];
    }
  }
}

// ---------------- aggregation ----------------
__global__ void k_self_scale(const float* __restrict__ h, const float* __restrict__ dinv,
                             float* __restrict__ agg, int d, int total) {
  int idx = blockIdx.x * blockDim.x + threadIdx.x;
  if (idx >= total) return;
  int i = idx / d;
  float s = dinv[i];
  agg[idx] = s * s * h[idx];
}
__global__ void k_self_bias(const float* __restrict__ h, const float* __restrict__ dinv,
                            const float* __restrict__ b, float* __restrict__ out,
                            int d, int strideH, int total) {
  int idx = blockIdx.x * blockDim.x + threadIdx.x;
  if (idx >= total) return;
  int i = idx / d, j = idx - i * d;
  float s = dinv[i];
  out[idx] = s * s * h[(size_t)i * strideH + j] + b[j];
}
__global__ void k_edge_agg(const long long* __restrict__ src, const long long* __restrict__ dst,
                           const float* __restrict__ dinv, const float* __restrict__ h,
                           float* __restrict__ out, int chunks, int strideH, int strideO,
                           int totalThreads) {
  int gid = blockIdx.x * blockDim.x + threadIdx.x;
  if (gid >= totalThreads) return;
  int e  = gid / chunks;
  int jb = (gid - e * chunks) * 4;
  int s = (int)src[e], t = (int)dst[e];
  float nrm = dinv[s] * dinv[t];
  float4 v = *(const float4*)(h + (size_t)s * strideH + jb);
  float* o = out + (size_t)t * strideO + jb;
  atomicAdd(o + 0, nrm * v.x);
  atomicAdd(o + 1, nrm * v.y);
  atomicAdd(o + 2, nrm * v.z);
  atomicAdd(o + 3, nrm * v.w);
}
__global__ void k_relu_bias_cvt(const float* __restrict__ agg, const float* __restrict__ b,
                                unsigned short* __restrict__ hbf, int d, int total) {
  int idx = blockIdx.x * blockDim.x + threadIdx.x;
  if (idx >= total) return;
  int j = idx % d;
  hbf[idx] = f2bf(fmaxf(agg[idx] + b[j], 0.0f));
}

extern "C" void kernel_launch(void* const* d_in, const int* in_sizes, int n_in,
                              void* d_out, int out_size, void* d_ws, size_t ws_size,
                              hipStream_t stream) {
  const float*     x   = (const float*)d_in[0];
  const long long* ei  = (const long long*)d_in[1];
  const float*     W1  = (const float*)d_in[2];
  const float*     b1  = (const float*)d_in[3];
  const float*     W2  = (const float*)d_in[4];
  const float*     b2  = (const float*)d_in[5];
  float*           out = (float*)d_out;

  const long long* srcE = ei;        // edge_index[0]
  const long long* dstE = ei + EE;   // edge_index[1]

  char* ws = (char*)d_ws;
  float*          deg  = (float*)(ws + OFF_DEG);
  float*          dinv = (float*)(ws + OFF_DINV);
  unsigned short* xbf  = (unsigned short*)(ws + OFF_XBF);
  unsigned short* wt1  = (unsigned short*)(ws + OFF_WT1);
  unsigned short* wt2  = (unsigned short*)(ws + OFF_WT2);
  unsigned short* hbf  = (unsigned short*)(ws + OFF_HBF);
  float*          h0   = (float*)(ws + OFF_H0);
  float*          agg1 = (float*)(ws + OFF_AGG1);
  float*          h2   = (float*)(ws + OFF_H2);

  const int B = 256;

  // ---- normalization ----
  k_init_deg<<<(NN + B - 1) / B, B, 0, stream>>>(deg, NN);
  k_deg<<<EE / B, B, 0, stream>>>(dstE, deg, EE);
  k_dinv<<<(NN + B - 1) / B, B, 0, stream>>>(deg, dinv, NN);

  // ---- bf16 casts / transposed weights ----
  {
    int tot = NN * KP1;
    k_cvt_pad<<<(tot + B - 1) / B, B, 0, stream>>>(x, xbf, DIN, KP1, tot);
  }
  {
    int tot = DMID * KP1;
    k_cvt_transpose<<<(tot + B - 1) / B, B, 0, stream>>>(W1, wt1, DIN, DMID, KP1, tot);
  }
  {
    int tot = NP2 * DMID;
    k_cvt_transpose<<<(tot + B - 1) / B, B, 0, stream>>>(W2, wt2, DMID, DIN, DMID, tot);
  }

  // ---- layer 1: h0 = x @ W1  (M=20000, N=512, K=2016) ----
  k_gemm_bf16<<<(NN / 32) * (DMID / 64) / 8, B, 0, stream>>>(xbf, wt1, h0, DMID, KP1, DMID / 64);
  {
    int tot = NN * DMID;
    k_self_scale<<<tot / B, B, 0, stream>>>(h0, dinv, agg1, DMID, tot);
  }
  {
    int chunks = DMID / 4, tot = EE * chunks;
    k_edge_agg<<<tot / B, B, 0, stream>>>(srcE, dstE, dinv, h0, agg1, chunks, DMID, DMID, tot);
  }
  {
    int tot = NN * DMID;
    k_relu_bias_cvt<<<tot / B, B, 0, stream>>>(agg1, b1, hbf, DMID, tot);
  }

  // ---- layer 2: h2 = h @ W2  (M=20000, N=2048(pad), K=512) ----
  k_gemm_bf16<<<(NN / 32) * (NP2 / 64) / 8, B, 0, stream>>>(hbf, wt2, h2, NP2, DMID, NP2 / 64);
  {
    int tot = NN * DIN;
    k_self_bias<<<(tot + B - 1) / B, B, 0, stream>>>(h2, dinv, b2, out, DIN, NP2, tot);
  }
  {
    int chunks = DIN / 4, tot = EE * chunks;
    k_edge_agg<<<tot / B, B, 0, stream>>>(srcE, dstE, dinv, h2, out, chunks, NP2, DIN, tot);
  }
}
